// Seq2seq_73349451481726
// MI455X (gfx1250) — matrix operand order repaired and checked by reading, not compiled, not run on hardware
//
#include <hip/hip_runtime.h>

#define H_    256
#define CIN_  16
#define TIN_  48
#define TOUT_ 24
#define COUT_ 8
#define KENC_ 288   // padded encoder K: [16 x | 256 h | 16 zero-pad]
#define KDEC_ 256

typedef __attribute__((ext_vector_type(16))) _Float16 v16h;
typedef __attribute__((ext_vector_type(8)))  _Float16 v8h;
typedef __attribute__((ext_vector_type(8)))  float    v8f;

#define GAS __attribute__((address_space(1)))
typedef GAS const _Float16* gch;   // global-AS f16 pointer
typedef GAS const v8h*      gv8h;  // global-AS 16B vector pointer

union V16 { v16h v; v8h h[2]; };

__device__ __forceinline__ v8f wmma_f16(const V16& a, const V16& b, v8f c) {
  // D = A(16x32 f16) * B(32x16 f16) + C(16x16 f32)
  return __builtin_amdgcn_wmma_f32_16x16x32_f16(false, a.v, false, b.v,
                                                (short)0, c, false, false);
}

// Launder the weight base through an empty volatile asm so LICM cannot prove
// the derived loads loop-invariant (prevents hoist-then-spill of ~288 VGPRs of
// B fragments out of the recurrence loop). "+s" keeps the base in an SGPR
// pair, and the addrspace(1) return type keeps the loads in GLOBAL_LOAD
// (saddr + voffset) form instead of FLAT (which would also burn DScnt).
__device__ __forceinline__ gch opaqueg(const _Float16* p) {
  unsigned long long v = (unsigned long long)p;
  asm volatile("" : "+s"(v));
  return (gch)v;
}

__device__ __forceinline__ float fsigm(float x) { return 1.0f / (1.0f + __expf(-x)); }
__device__ __forceinline__ float ftanh(float x) { return 1.0f - 2.0f / (__expf(2.0f * x) + 1.0f); }

// ---------------- weight repack: f32 -> row-major f16, padded ----------------
__global__ void prep_weights(const float* __restrict__ Wih,
                             const float* __restrict__ WhhE,
                             const float* __restrict__ WhhD,
                             const float* __restrict__ denseW,
                             _Float16* __restrict__ wenc,
                             _Float16* __restrict__ wdec,
                             _Float16* __restrict__ wdns) {
  int i = blockIdx.x * blockDim.x + threadIdx.x;
  if (i < 1024 * KENC_) {               // encoder: [Wih(16) | Whh(256) | 0(16)]
    int r = i / KENC_, c = i % KENC_;
    float v = 0.0f;
    if (c < CIN_)            v = Wih[r * CIN_ + c];
    else if (c < CIN_ + H_)  v = WhhE[r * H_ + (c - CIN_)];
    wenc[i] = (_Float16)v;
  }
  if (i < 1024 * KDEC_) wdec[i] = (_Float16)WhhD[i];
  if (i < TOUT_ * 16 * H_) {            // dense: pad C_OUT 8 -> 16 rows
    int t = i / (16 * H_), rem = i % (16 * H_), o = rem / H_, k = rem % H_;
    wdns[i] = (_Float16)((o < COUT_) ? denseW[(t * COUT_ + o) * H_ + k] : 0.0f);
  }
}

// ---------------- main fused seq2seq kernel ----------------
// grid: 256 blocks x 512 threads (16 waves). Block owns 32 batch rows.
// wave w owns H-columns [w*16, w*16+16) across all 4 gates; c state in regs.
__global__ __launch_bounds__(512) void seq2seq_kernel(
    const float* __restrict__ x,
    const _Float16* __restrict__ wenc,
    const _Float16* __restrict__ wdec,
    const _Float16* __restrict__ wdns,
    const float* __restrict__ enc_b,
    const float* __restrict__ dec_b,
    const float* __restrict__ dense_b,
    float* __restrict__ out) {
  __shared__ __align__(16) _Float16 hbuf[2][32 * KENC_];

  const int tid  = threadIdx.x;
  const int lane = tid & 31;
  const int wv   = tid >> 5;       // wave id 0..15 -> H-column tile
  const int n16  = lane & 15;
  const int hi   = lane >> 4;
  const int row0 = blockIdx.x * 32;

  // zero both buffers: initial h = 0, pad columns stay 0 forever
  for (int i = tid; i < 2 * 32 * KENC_; i += 512) ((_Float16*)hbuf)[i] = (_Float16)0;

  float be[4], bd[4];
#pragma unroll
  for (int g = 0; g < 4; ++g) {
    be[g] = enc_b[g * H_ + wv * 16 + n16];
    bd[g] = dec_b[g * H_ + wv * 16 + n16];
  }

  v8f cst[2]; cst[0] = {}; cst[1] = {};
  __syncthreads();

  int p = 0;
  // ===================== Encoder: 48 steps =====================
  for (int t = 0; t < TIN_; ++t) {
    { // stage x_t (32 rows x 16 ch) into cols [0,16) of read buffer
      int r = tid >> 4, c = tid & 15;
      hbuf[p][r * KENC_ + c] =
          (_Float16)x[(size_t)(row0 + r) * (CIN_ * TIN_) + c * TIN_ + t];
    }
    __syncthreads();

    gch we = opaqueg(wenc);   // per-iteration opaque global base: no hoist

    v8f acc[4][2];
#pragma unroll
    for (int g = 0; g < 4; ++g) { acc[g][0] = {}; acc[g][1] = {}; }

#pragma unroll
    for (int kc = 0; kc < 9; ++kc) {
      const int ka = kc * 32 + 8 * hi;    // A-frag K offset (interleaved layout)
      const int kb = kc * 32 + 8 * hi;    // B-frag K offset (interleaved layout)
      V16 a[2];
#pragma unroll
      for (int rt = 0; rt < 2; ++rt) {
        const _Float16* ap = &hbuf[p][(rt * 16 + n16) * KENC_ + ka];
        a[rt].h[0] = *(const v8h*)(ap);
        a[rt].h[1] = *(const v8h*)(ap + 16);
      }
#pragma unroll
      for (int g = 0; g < 4; ++g) {
        gch bp = we + (size_t)(g * H_ + wv * 16 + n16) * KENC_ + kb;
        V16 b;
        b.h[0] = *(gv8h)(bp);
        b.h[1] = *(gv8h)(bp + 16);
        acc[g][0] = wmma_f16(a[0], b, acc[g][0]);
        acc[g][1] = wmma_f16(a[1], b, acc[g][1]);
      }
    }

    const int q = 1 - p;
#pragma unroll
    for (int rt = 0; rt < 2; ++rt) {
#pragma unroll
      for (int e = 0; e < 8; ++e) {       // D element e -> row m = e + 8*hi
        float ig = fsigm(acc[0][rt][e] + be[0]);
        float fg = fsigm(acc[1][rt][e] + be[1]);
        float gg = ftanh(acc[2][rt][e] + be[2]);
        float og = fsigm(acc[3][rt][e] + be[3]);
        float cn = fg * cst[rt][e] + ig * gg;
        cst[rt][e] = cn;
        float hn = og * ftanh(cn);
        hbuf[q][(rt * 16 + 8 * hi + e) * KENC_ + (CIN_ + wv * 16 + n16)] = (_Float16)hn;
      }
    }
    __syncthreads();
    p = q;
  }

  // ===================== Decoder: 24 steps (c resets to 0) =====================
  cst[0] = {}; cst[1] = {};
  for (int t = 0; t < TOUT_; ++t) {
    gch wd = opaqueg(wdec);   // per-iteration opaque global base: no hoist

    v8f acc[4][2];
#pragma unroll
    for (int g = 0; g < 4; ++g) { acc[g][0] = {}; acc[g][1] = {}; }

#pragma unroll
    for (int kc = 0; kc < 8; ++kc) {
      const int ka = kc * 32 + 8 * hi;
      const int kb = kc * 32 + 8 * hi;
      V16 a[2];
#pragma unroll
      for (int rt = 0; rt < 2; ++rt) {
        const _Float16* ap = &hbuf[p][(rt * 16 + n16) * KENC_ + CIN_ + ka];
        a[rt].h[0] = *(const v8h*)(ap);
        a[rt].h[1] = *(const v8h*)(ap + 16);
      }
#pragma unroll
      for (int g = 0; g < 4; ++g) {
        gch bp = wd + (size_t)(g * H_ + wv * 16 + n16) * KDEC_ + kb;
        V16 b;
        b.h[0] = *(gv8h)(bp);
        b.h[1] = *(gv8h)(bp + 16);
        acc[g][0] = wmma_f16(a[0], b, acc[g][0]);
        acc[g][1] = wmma_f16(a[1], b, acc[g][1]);
      }
    }

    const int q = 1 - p;
#pragma unroll
    for (int rt = 0; rt < 2; ++rt) {
#pragma unroll
      for (int e = 0; e < 8; ++e) {
        float ig = fsigm(acc[0][rt][e] + bd[0]);
        float fg = fsigm(acc[1][rt][e] + bd[1]);
        float gg = ftanh(acc[2][rt][e] + bd[2]);
        float og = fsigm(acc[3][rt][e] + bd[3]);
        float cn = fg * cst[rt][e] + ig * gg;
        cst[rt][e] = cn;
        float hn = og * ftanh(cn);
        hbuf[q][(rt * 16 + 8 * hi + e) * KENC_ + (CIN_ + wv * 16 + n16)] = (_Float16)hn;
      }
    }
    __syncthreads();   // h_t complete in hbuf[q]

    // dense: out[:, :, t] = h_t @ denseW[t]^T + dense_b[t]; waves 0,1 (rt = wv)
    if (wv < 2) {
      const int rt = wv;
      gch wn = opaqueg(wdns);
      v8f da = {};
#pragma unroll
      for (int kc = 0; kc < 8; ++kc) {
        const int ka = kc * 32 + 8 * hi;
        const int kb = kc * 32 + 8 * hi;
        V16 a, b;
        const _Float16* ap = &hbuf[q][(rt * 16 + n16) * KENC_ + CIN_ + ka];
        a.h[0] = *(const v8h*)(ap);
        a.h[1] = *(const v8h*)(ap + 16);
        gch bp = wn + (size_t)(t * 16 + n16) * H_ + kb;
        b.h[0] = *(gv8h)(bp);
        b.h[1] = *(gv8h)(bp + 16);
        da = wmma_f16(a, b, da);
      }
      if (n16 < COUT_) {
        float db = dense_b[t * COUT_ + n16];
#pragma unroll
        for (int e = 0; e < 8; ++e) {
          int gr = row0 + rt * 16 + 8 * hi + e;
          out[(size_t)gr * (COUT_ * TOUT_) + n16 * TOUT_ + t] = da[e] + db;
        }
      }
    }
    p = q;
  }
}

extern "C" void kernel_launch(void* const* d_in, const int* in_sizes, int n_in,
                              void* d_out, int out_size, void* d_ws, size_t ws_size,
                              hipStream_t stream) {
  const float* x       = (const float*)d_in[0];
  const float* enc_Wih = (const float*)d_in[1];
  const float* enc_Whh = (const float*)d_in[2];
  const float* enc_b   = (const float*)d_in[3];
  const float* dec_Whh = (const float*)d_in[4];
  const float* dec_b   = (const float*)d_in[5];
  const float* dense_W = (const float*)d_in[6];
  const float* dense_b = (const float*)d_in[7];
  float* out = (float*)d_out;

  _Float16* wenc = (_Float16*)d_ws;                 // 1024 x 288
  _Float16* wdec = wenc + 1024 * KENC_;             // 1024 x 256
  _Float16* wdns = wdec + 1024 * KDEC_;             // 24 x 16 x 256

  const int prep_n = 1024 * KENC_;                  // covers all three ranges
  prep_weights<<<(prep_n + 255) / 256, 256, 0, stream>>>(
      enc_Wih, enc_Whh, dec_Whh, dense_W, wenc, wdec, wdns);

  seq2seq_kernel<<<8192 / 32, 512, 0, stream>>>(
      x, wenc, wdec, wdns, enc_b, dec_b, dense_b, out);
}
